// LSTMCovid_53274774340275
// MI455X (gfx1250) — matrix operand order repaired and checked by reading, not compiled, not run on hardware
//
#include <hip/hip_runtime.h>
#include <hip/hip_bf16.h>

// ---------------------------------------------------------------------------
// 2-layer LSTM (H=128) over B=1024, T=512, F=32 + small FC head, for MI455X.
// Persistent per-batch-tile recurrence kernels: combined [W_hh | W_ih] weights
// in LDS as f16 (CDNA5 320KB/WGP), gates via v_wmma_f32_16x16x32_f16 with the
// bias folded into the C fragment and the input projection fused into the
// recurrence (A = [h_{t-1} | x_t]). c-state lives in VGPRs. B-fragment and
// A-fragment LDS loads are software-pipelined one WMMA ahead; the next step's
// input slice is prefetched into registers at the top of each step.
// ---------------------------------------------------------------------------

typedef __attribute__((ext_vector_type(16))) _Float16 v16h;
typedef __attribute__((ext_vector_type(8)))  float    v8f;

#define T_STEPS 512
#define BATCH   1024
#define HID     128
#define G4      512      // 4*HID (gate rows, PyTorch order i,f,g,o)
#define BT      64       // batch rows per workgroup
#define NWG     (BATCH / BT)

// LDS strides (halves). +16 halves = 32B = 8-bank skew between consecutive
// rows so b128 fragment loads don't all land on the same banks.
#define L0_LDW 176       // layer0 combined weight row: 160 cols used
#define L0_LDA 176       // layer0 A buffer row: 160 cols used ([h(128)|x(32)])
#define L1_LDW 272       // layer1 combined weight row: 256 cols used
#define L1_LDA 272       // layer1 A buffer row: 256 cols used ([h(128)|x(128)])

// ---- WMMA fragment loaders (wave32, 16x16x32 f16) -------------------------

// A fragment: 16x32 f16 tile from row-major src (row stride ld in halves).
// ISA layout: lanes 0-15 row M=lane, e0..7 -> K=0..7, e8..15 -> K=16..23;
// lanes 16-31 same rows, K shifted by +8 (K=8..15 / 24..31).
__device__ __forceinline__ v16h load_a16x32(const _Float16* src, int ld) {
  const int lane = threadIdx.x & 31;
  const int row  = lane & 15;
  const int kh   = (lane >> 4) & 1;
  const _Float16* p = src + row * ld + kh * 8;
  v16h a;
#pragma unroll
  for (int e = 0; e < 8; ++e) a[e] = p[e];          // one 16B load
#pragma unroll
  for (int e = 0; e < 8; ++e) a[8 + e] = p[16 + e]; // one 16B load
  return a;
}

// B fragment: 32x16 where B[k][n] = W[n0+n][k] (W row-major, stride ld).
// Lanes 0-15 hold column N=lane with e0..7 -> K=0..7, e8..15 -> K=16..23;
// lanes 16-31 K shifted by +8 (same order as the A fragment).
__device__ __forceinline__ v16h load_bT32x16(const _Float16* w, int ld) {
  const int lane = threadIdx.x & 31;
  const int n    = lane & 15;
  const int kb   = ((lane >> 4) & 1) * 8; // 0 or 8
  const _Float16* p = w + n * ld + kb;
  v16h b;
#pragma unroll
  for (int e = 0; e < 8; ++e) b[e] = p[e];          // one 16B load
#pragma unroll
  for (int e = 0; e < 8; ++e) b[8 + e] = p[16 + e]; // one 16B load
  return b;
}

__device__ __forceinline__ float sigmoidf_(float x) {
  return 1.0f / (1.0f + __expf(-x));
}

// Software-pipelined gate GEMM for a pair of h-column positions.
// acc[p2][g] += A(bt, kb) x B(g, hcbase+p2, kb) over kb = 0..KB-1.
// Every WMMA's B fragment is loaded one WMMA earlier; A one kb earlier.
template <int KB, int LDW, int LDA>
__device__ __forceinline__ void pair_gemm(const _Float16* __restrict__ Wc,
                                          const _Float16* __restrict__ abase,
                                          int hcbase, v8f acc[2][4]) {
  auto bp = [&](int u, int kb) -> const _Float16* {
    const int p2 = u >> 2, g = u & 3;
    return Wc + (g * HID + (hcbase + p2) * 16) * LDW + kb * 32;
  };
  v16h A = load_a16x32(abase, LDA);
  v16h B = load_bT32x16(bp(0, 0), LDW);
#pragma unroll
  for (int kb = 0; kb < KB; ++kb) {
    v16h An;
    if (kb + 1 < KB) An = load_a16x32(abase + (kb + 1) * 32, LDA);
#pragma unroll
    for (int u = 0; u < 8; ++u) {
      v16h Bn;
      if (u < 7)            Bn = load_bT32x16(bp(u + 1, kb), LDW);
      else if (kb + 1 < KB) Bn = load_bT32x16(bp(0, kb + 1), LDW);
      acc[u >> 2][u & 3] = __builtin_amdgcn_wmma_f32_16x16x32_f16(
          false, A, false, B, (short)0, acc[u >> 2][u & 3], false, false);
      if (u < 7 || kb + 1 < KB) B = Bn;
    }
    if (kb + 1 < KB) A = An;
  }
}

// ---- weight conversion f32 -> f16 -----------------------------------------
__global__ void cvt_weights(const float* __restrict__ wih0, const float* __restrict__ whh0,
                            const float* __restrict__ wih1, const float* __restrict__ whh1,
                            _Float16* __restrict__ o_ih0, _Float16* __restrict__ o_hh0,
                            _Float16* __restrict__ o_ih1, _Float16* __restrict__ o_hh1) {
  int i = blockIdx.x * blockDim.x + threadIdx.x;
  if (i < G4 * 32) o_ih0[i] = (_Float16)wih0[i];
  if (i < G4 * HID) {
    o_hh0[i] = (_Float16)whh0[i];
    o_ih1[i] = (_Float16)wih1[i];
    o_hh1[i] = (_Float16)whh1[i];
  }
}

// ---- LSTM layer 0: x f32 [B,T,32] -> hs f16 [T,B,128] ---------------------
__launch_bounds__(256, 1)
__global__ void lstm_layer0(const float* __restrict__ x,
                            const _Float16* __restrict__ wih,  // [512,32] f16
                            const _Float16* __restrict__ whh,  // [512,128] f16
                            const float* __restrict__ bih, const float* __restrict__ bhh,
                            _Float16* __restrict__ hs_out) {   // [T,B,128] f16
  extern __shared__ _Float16 smem[];
  _Float16* Wc   = smem;                 // [512][L0_LDW]: 0..127 = W_hh, 128..159 = W_ih
  _Float16* abuf = smem + G4 * L0_LDW;   // [64][L0_LDA]:  0..127 = h_{t-1}, 128..159 = x_t

  const int tid = threadIdx.x;
  const int b0  = blockIdx.x * BT;

  for (int i = tid; i < G4 * HID; i += 256)
    Wc[(i >> 7) * L0_LDW + (i & 127)] = whh[i];
  for (int i = tid; i < G4 * 32; i += 256)
    Wc[(i >> 5) * L0_LDW + 128 + (i & 31)] = wih[i];
  for (int i = tid; i < BT * HID; i += 256)
    abuf[(i >> 7) * L0_LDA + (i & 127)] = (_Float16)0.0f;  // h_{-1} = 0
  for (int i = tid; i < BT * 32; i += 256) {
    int r = i >> 5, c = i & 31;
    abuf[r * L0_LDA + 128 + c] = (_Float16)x[((size_t)(b0 + r) * T_STEPS) * 32 + c];
  }

  const int wv   = tid >> 5;
  const int lane = tid & 31;
  const int bt   = wv >> 1;        // 0..3   batch sub-tile (16 rows)
  const int hc0  = (wv & 1) * 4;   // h-column tile base (0 or 4)
  const int nn   = lane & 15;      // N within tile
  const int mhi  = (lane >> 4) * 8;// M base of this lane's 8 accumulator rows

  // per-thread coalesced staging assignment: x slice (8 f32) and hs copy (32 f16)
  const int xr = (tid * 8) >> 5,  xc = (tid * 8) & 31;    // x: [64][32]
  const int hr = (tid * 32) >> 7, hcc = (tid * 32) & 127; // h: [64][128]

  float bias[4][4];                // [position][gate]
#pragma unroll
  for (int p = 0; p < 4; ++p)
#pragma unroll
    for (int g = 0; g < 4; ++g) {
      int col = g * HID + (hc0 + p) * 16 + nn;
      bias[p][g] = bih[col] + bhh[col];
    }

  v8f cst[4];
#pragma unroll
  for (int p = 0; p < 4; ++p)
#pragma unroll
    for (int e = 0; e < 8; ++e) cst[p][e] = 0.0f;

  __syncthreads();

  for (int t = 0; t < T_STEPS; ++t) {
    // prefetch next step's x slice into registers (hides under WMMA phase)
    float xreg[8];
    if (t + 1 < T_STEPS) {
      const float* src = x + ((size_t)(b0 + xr) * T_STEPS + (t + 1)) * 32 + xc;
#pragma unroll
      for (int e = 0; e < 8; ++e) xreg[e] = src[e];
    }

    _Float16 hval[4][8];
#pragma unroll
    for (int ph = 0; ph < 2; ++ph) {           // two position-pairs
      const int hcb = hc0 + ph * 2;
      v8f acc[2][4];
#pragma unroll
      for (int p2 = 0; p2 < 2; ++p2)
#pragma unroll
        for (int g = 0; g < 4; ++g)
#pragma unroll
          for (int e = 0; e < 8; ++e) acc[p2][g][e] = bias[ph * 2 + p2][g];

      pair_gemm<5, L0_LDW, L0_LDA>(Wc, &abuf[(bt * 16) * L0_LDA], hcb, acc);

#pragma unroll
      for (int p2 = 0; p2 < 2; ++p2) {
        const int p = ph * 2 + p2;
#pragma unroll
        for (int e = 0; e < 8; ++e) {
          float ig = sigmoidf_(acc[p2][0][e]);
          float fg = sigmoidf_(acc[p2][1][e]);
          float gg = tanhf(acc[p2][2][e]);
          float og = sigmoidf_(acc[p2][3][e]);
          float cv = fg * cst[p][e] + ig * gg;
          cst[p][e] = cv;
          hval[p][e] = (_Float16)(og * tanhf(cv));
        }
      }
    }
    __syncthreads();  // all reads of abuf for step t complete
#pragma unroll
    for (int p = 0; p < 4; ++p) {
      _Float16* dst = &abuf[(bt * 16) * L0_LDA + (hc0 + p) * 16 + nn];
#pragma unroll
      for (int e = 0; e < 8; ++e) dst[(mhi + e) * L0_LDA] = hval[p][e];
    }
    if (t + 1 < T_STEPS) {
#pragma unroll
      for (int e = 0; e < 8; ++e)
        abuf[xr * L0_LDA + 128 + xc + e] = (_Float16)xreg[e];
    }
    __syncthreads();  // abuf holds h_t | x_{t+1}
    // coalesced h_t stream-out: LDS -> global (b128s), overlaps next compute
    {
      _Float16* dst = hs_out + ((size_t)t * BATCH + b0 + hr) * HID + hcc;
      const _Float16* srcl = &abuf[hr * L0_LDA + hcc];
#pragma unroll
      for (int e = 0; e < 32; ++e) dst[e] = srcl[e];
    }
  }
}

// ---- LSTM layer 1: hs f16 [T,B,128] -> hT f32 [B,128] ---------------------
__launch_bounds__(256, 1)
__global__ void lstm_layer1(const _Float16* __restrict__ hs_in, // [T,B,128] f16
                            const _Float16* __restrict__ wih,   // [512,128] f16
                            const _Float16* __restrict__ whh,   // [512,128] f16
                            const float* __restrict__ bih, const float* __restrict__ bhh,
                            float* __restrict__ hT_out) {       // [B,128] f32
  extern __shared__ _Float16 smem[];
  _Float16* Wc   = smem;                 // [512][L1_LDW]: 0..127 = W_hh, 128..255 = W_ih
  _Float16* abuf = smem + G4 * L1_LDW;   // [64][L1_LDA]:  0..127 = h_{t-1}, 128..255 = x_t

  const int tid = threadIdx.x;
  const int b0  = blockIdx.x * BT;

  for (int i = tid; i < G4 * HID; i += 256) {
    int r = i >> 7, c = i & 127;
    Wc[r * L1_LDW + c]       = whh[i];
    Wc[r * L1_LDW + 128 + c] = wih[i];
  }
  for (int i = tid; i < BT * HID; i += 256)
    abuf[(i >> 7) * L1_LDA + (i & 127)] = (_Float16)0.0f;  // h_{-1} = 0
  for (int i = tid; i < BT * HID; i += 256) {
    int r = i >> 7, c = i & 127;
    abuf[r * L1_LDA + 128 + c] = hs_in[(size_t)(b0 + r) * HID + c];  // x_0
  }

  const int wv   = tid >> 5;
  const int lane = tid & 31;
  const int bt   = wv >> 1;
  const int hc0  = (wv & 1) * 4;
  const int nn   = lane & 15;
  const int mhi  = (lane >> 4) * 8;

  const int hr = (tid * 32) >> 7, hcc = (tid * 32) & 127; // staging: [64][128]

  float bias[4][4];
#pragma unroll
  for (int p = 0; p < 4; ++p)
#pragma unroll
    for (int g = 0; g < 4; ++g) {
      int col = g * HID + (hc0 + p) * 16 + nn;
      bias[p][g] = bih[col] + bhh[col];
    }

  v8f cst[4];
#pragma unroll
  for (int p = 0; p < 4; ++p)
#pragma unroll
    for (int e = 0; e < 8; ++e) cst[p][e] = 0.0f;

  __syncthreads();

  for (int t = 0; t < T_STEPS; ++t) {
    // prefetch next step's hs0 slice (coalesced b128 loads, L2-resident)
    _Float16 xreg[32];
    if (t + 1 < T_STEPS) {
      const _Float16* src = hs_in + ((size_t)(t + 1) * BATCH + b0 + hr) * HID + hcc;
#pragma unroll
      for (int e = 0; e < 32; ++e) xreg[e] = src[e];
    }

    _Float16 hval[4][8];
#pragma unroll
    for (int ph = 0; ph < 2; ++ph) {
      const int hcb = hc0 + ph * 2;
      v8f acc[2][4];
#pragma unroll
      for (int p2 = 0; p2 < 2; ++p2)
#pragma unroll
        for (int g = 0; g < 4; ++g)
#pragma unroll
          for (int e = 0; e < 8; ++e) acc[p2][g][e] = bias[ph * 2 + p2][g];

      pair_gemm<8, L1_LDW, L1_LDA>(Wc, &abuf[(bt * 16) * L1_LDA], hcb, acc);

#pragma unroll
      for (int p2 = 0; p2 < 2; ++p2) {
        const int p = ph * 2 + p2;
#pragma unroll
        for (int e = 0; e < 8; ++e) {
          float ig = sigmoidf_(acc[p2][0][e]);
          float fg = sigmoidf_(acc[p2][1][e]);
          float gg = tanhf(acc[p2][2][e]);
          float og = sigmoidf_(acc[p2][3][e]);
          float cv = fg * cst[p][e] + ig * gg;
          cst[p][e] = cv;
          float hv = og * tanhf(cv);
          hval[p][e] = (_Float16)hv;
          if (t == T_STEPS - 1) {
            hT_out[(size_t)(b0 + bt * 16 + mhi + e) * HID + (hc0 + p) * 16 + nn] = hv;
          }
        }
      }
    }
    __syncthreads();
#pragma unroll
    for (int p = 0; p < 4; ++p) {
      _Float16* dst = &abuf[(bt * 16) * L1_LDA + (hc0 + p) * 16 + nn];
#pragma unroll
      for (int e = 0; e < 8; ++e) dst[(mhi + e) * L1_LDA] = hval[p][e];
    }
    if (t + 1 < T_STEPS) {
#pragma unroll
      for (int e = 0; e < 32; ++e)
        abuf[hr * L1_LDA + 128 + hcc + e] = xreg[e];
    }
    __syncthreads();
  }
}

// ---- FC head (negligible FLOPs; plain VALU) -------------------------------
__global__ void fc1_kernel(const float* __restrict__ hT, const float* __restrict__ W1,
                           const float* __restrict__ b1, float* __restrict__ a1) {
  int i = blockIdx.x * blockDim.x + threadIdx.x;  // over B*64
  int b = i >> 6, j = i & 63;
  const float* h = hT + (size_t)b * 128;
  const float* w = W1 + (size_t)j * 128;
  float s = b1[j];
#pragma unroll 4
  for (int k = 0; k < 128; ++k) s = fmaf(h[k], w[k], s);
  a1[i] = fmaxf(s, 0.0f);
}

__global__ void fc2_kernel(const float* __restrict__ a1, const float* __restrict__ W2,
                           const float* __restrict__ b2, float* __restrict__ out) {
  int b = blockIdx.x * blockDim.x + threadIdx.x;  // over B
  const float* a = a1 + (size_t)b * 64;
  float s = b2[0];
#pragma unroll 4
  for (int k = 0; k < 64; ++k) s = fmaf(a[k], W2[k], s);
  out[b] = fmaxf(s, 0.0f);
}

// ---------------------------------------------------------------------------
extern "C" void kernel_launch(void* const* d_in, const int* in_sizes, int n_in,
                              void* d_out, int out_size, void* d_ws, size_t ws_size,
                              hipStream_t stream) {
  const float* x     = (const float*)d_in[0];
  const float* W_ih0 = (const float*)d_in[1];
  const float* W_hh0 = (const float*)d_in[2];
  const float* b_ih0 = (const float*)d_in[3];
  const float* b_hh0 = (const float*)d_in[4];
  const float* W_ih1 = (const float*)d_in[5];
  const float* W_hh1 = (const float*)d_in[6];
  const float* b_ih1 = (const float*)d_in[7];
  const float* b_hh1 = (const float*)d_in[8];
  const float* W1    = (const float*)d_in[9];
  const float* b1    = (const float*)d_in[10];
  const float* W2    = (const float*)d_in[11];
  const float* b2    = (const float*)d_in[12];

  // Workspace carve-up (~135.3 MB total; hs0 is the 128 MB inter-layer tensor)
  char* ws = (char*)d_ws;
  size_t off = 0;
  auto carve = [&](size_t bytes) -> void* {
    void* p = ws + off;
    off = (off + bytes + 255) & ~(size_t)255;
    return p;
  };
  _Float16* wih0h = (_Float16*)carve((size_t)G4 * 32 * 2);
  _Float16* whh0h = (_Float16*)carve((size_t)G4 * HID * 2);
  _Float16* wih1h = (_Float16*)carve((size_t)G4 * HID * 2);
  _Float16* whh1h = (_Float16*)carve((size_t)G4 * HID * 2);
  _Float16* hs0   = (_Float16*)carve((size_t)T_STEPS * BATCH * HID * 2);
  float*    hT1   = (float*)carve((size_t)BATCH * HID * 4);
  float*    a1    = (float*)carve((size_t)BATCH * 64 * 4);

  cvt_weights<<<(G4 * HID + 255) / 256, 256, 0, stream>>>(
      W_ih0, W_hh0, W_ih1, W_hh1, wih0h, whh0h, wih1h, whh1h);

  const size_t lds0 = (size_t)(G4 * L0_LDW + BT * L0_LDA) * sizeof(_Float16); // ~198 KB
  lstm_layer0<<<NWG, 256, lds0, stream>>>(x, wih0h, whh0h, b_ih0, b_hh0, hs0);

  const size_t lds1 = (size_t)(G4 * L1_LDW + BT * L1_LDA) * sizeof(_Float16); // ~306 KB
  lstm_layer1<<<NWG, 256, lds1, stream>>>(hs0, wih1h, whh1h, b_ih1, b_hh1, hT1);

  fc1_kernel<<<(BATCH * 64) / 256, 256, 0, stream>>>(hT1, W1, b1, a1);
  fc2_kernel<<<BATCH / 256, 256, 0, stream>>>(a1, W2, b2, (float*)d_out);
}
